// Linear8bit_19043884991169
// MI455X (gfx1250) — 8-bit matrix operand order repaired and checked by reading, compiled, not run on hardware
//
#include <hip/hip_runtime.h>
#include <stdint.h>

#define DIM_M 8192
#define DIM_N 4096
#define DIM_K 4096

typedef int v8i __attribute__((ext_vector_type(8)));

__device__ __forceinline__ unsigned int pack4(float a, float b, float c, float d,
                                              float inv) {
    int q0 = __float2int_rn(a * inv);
    int q1 = __float2int_rn(b * inv);
    int q2 = __float2int_rn(c * inv);
    int q3 = __float2int_rn(d * inv);
    q0 = min(127, max(-127, q0));
    q1 = min(127, max(-127, q1));
    q2 = min(127, max(-127, q2));
    q3 = min(127, max(-127, q3));
    return (unsigned int)(q0 & 255) | ((unsigned int)(q1 & 255) << 8) |
           ((unsigned int)(q2 & 255) << 16) | ((unsigned int)(q3 & 255) << 24);
}

__global__ __launch_bounds__(256)
void quant_rows_kernel(const float* __restrict__ src,
                       int8_t* __restrict__ dst,
                       float* __restrict__ scales) {
    const int row = blockIdx.x;
    const int tid = threadIdx.x;

    const float4* rp = (const float4*)(src + (size_t)row * DIM_K) + tid * 4;
    float4 v0 = rp[0];
    float4 v1 = rp[1];
    float4 v2 = rp[2];
    float4 v3 = rp[3];

    float m = 0.0f;
    m = fmaxf(m, fmaxf(fmaxf(fabsf(v0.x), fabsf(v0.y)), fmaxf(fabsf(v0.z), fabsf(v0.w))));
    m = fmaxf(m, fmaxf(fmaxf(fabsf(v1.x), fabsf(v1.y)), fmaxf(fabsf(v1.z), fabsf(v1.w))));
    m = fmaxf(m, fmaxf(fmaxf(fabsf(v2.x), fabsf(v2.y)), fmaxf(fabsf(v2.z), fabsf(v2.w))));
    m = fmaxf(m, fmaxf(fmaxf(fabsf(v3.x), fabsf(v3.y)), fmaxf(fabsf(v3.z), fabsf(v3.w))));

    __shared__ float red[256];
    red[tid] = m;
    __syncthreads();
#pragma unroll
    for (int off = 128; off > 0; off >>= 1) {
        if (tid < off) red[tid] = fmaxf(red[tid], red[tid + off]);
        __syncthreads();
    }
    const float s = fmaxf(red[0], 1e-30f);
    const float inv = 127.0f / s;

    uint4 packed;
    packed.x = pack4(v0.x, v0.y, v0.z, v0.w, inv);
    packed.y = pack4(v1.x, v1.y, v1.z, v1.w, inv);
    packed.z = pack4(v2.x, v2.y, v2.z, v2.w, inv);
    packed.w = pack4(v3.x, v3.y, v3.z, v3.w, inv);
    *(uint4*)(dst + (size_t)row * DIM_K + (size_t)tid * 16) = packed;

    if (tid == 0) scales[row] = s;
}

#define TM 128
#define TN 128
#define TK 64
#define LDP 80

__global__ __launch_bounds__(256)
void igemm_wmma_kernel(const int8_t* __restrict__ A,
                       const int8_t* __restrict__ B,
                       const float* __restrict__ s2,
                       const float* __restrict__ s1,
                       const float* __restrict__ bias,
                       float* __restrict__ out) {
    __shared__ int8_t lA[TM * LDP];
    __shared__ int8_t lB[TN * LDP];

    const int tid  = threadIdx.x;
    const int wave = tid >> 5;
    const int lane = tid & 31;
    const int tileM = blockIdx.y * TM;
    const int tileN = blockIdx.x * TN;

    v8i acc[8] = {};

    const int ldRow = tid >> 1;
    const int ldOff = (tid & 1) * 32;
    const int8_t* gA = A + (size_t)(tileM + ldRow) * DIM_K + ldOff;
    const int8_t* gB = B + (size_t)(tileN + ldRow) * DIM_K + ldOff;

    const int aRow  = wave * 16 + (lane & 15);
    const int aBase = (lane < 16) ? 0 : 16;
    const int kHalf = (lane < 16) ? 0 : 16;
    const int8_t* apL = &lA[aRow * LDP + aBase];

    for (int k0 = 0; k0 < DIM_K; k0 += TK) {
        uint4 a0 = *(const uint4*)(gA + k0);
        uint4 a1 = *(const uint4*)(gA + k0 + 16);
        uint4 b0 = *(const uint4*)(gB + k0);
        uint4 b1 = *(const uint4*)(gB + k0 + 16);
        *(uint4*)(&lA[ldRow * LDP + ldOff])      = a0;
        *(uint4*)(&lA[ldRow * LDP + ldOff + 16]) = a1;
        *(uint4*)(&lB[ldRow * LDP + ldOff])      = b0;
        *(uint4*)(&lB[ldRow * LDP + ldOff + 16]) = b1;

        if (k0 + TK < DIM_K) {
            __builtin_prefetch(gA + k0 + TK, 0, 0);
            __builtin_prefetch(gB + k0 + TK, 0, 0);
        }
        __syncthreads();

        v8i a;
        {
            int2 p0 = *(const int2*)(apL + 0);
            int2 p1 = *(const int2*)(apL + 8);
            int2 p2 = *(const int2*)(apL + 32);
            int2 p3 = *(const int2*)(apL + 40);
            a[0] = p0.x; a[1] = p0.y;
            a[2] = p1.x; a[3] = p1.y;
            a[4] = p2.x; a[5] = p2.y;
            a[6] = p3.x; a[7] = p3.y;
        }

#pragma unroll
        for (int n = 0; n < 8; ++n) {
            const int bCol = n * 16 + (lane & 15);
            const int8_t* bp = &lB[bCol * LDP + kHalf];
            int4 q0 = *(const int4*)(bp);
            int4 q1 = *(const int4*)(bp + 32);
            v8i b;
            b[0] = q0.x; b[1] = q0.y; b[2] = q0.z; b[3] = q0.w;
            b[4] = q1.x; b[5] = q1.y; b[6] = q1.z; b[7] = q1.w;
            acc[n] = __builtin_amdgcn_wmma_i32_16x16x64_iu8(true, a, true, b,
                                                            acc[n], false, false);
        }
        __syncthreads();
    }

    const int colLane = lane & 15;
    const int rowAdd  = (lane < 16) ? 0 : 8;
    const int mBase   = tileM + wave * 16 + rowAdd;

    float sc2v[8];
#pragma unroll
    for (int j = 0; j < 8; ++j) sc2v[j] = s2[mBase + j] * (1.0f / 127.0f);

#pragma unroll
    for (int n = 0; n < 8; ++n) {
        const int col   = tileN + n * 16 + colLane;
        const float sc1 = s1[col] * (1.0f / 127.0f);
        const float bv  = bias[col];
#pragma unroll
        for (int j = 0; j < 8; ++j) {
            out[(size_t)(mBase + j) * DIM_N + col] =
                (float)acc[n][j] * sc1 * sc2v[j] + bv;
        }
    }
}

extern "C" void kernel_launch(void* const* d_in, const int* in_sizes, int n_in,
                              void* d_out, int out_size, void* d_ws, size_t ws_size,
                              hipStream_t stream) {
    (void)in_sizes; (void)n_in; (void)out_size; (void)ws_size;

    const float* x    = (const float*)d_in[0];
    const float* w    = (const float*)d_in[1];
    const float* bias = (const float*)d_in[2];
    float* out = (float*)d_out;

    int8_t* x8 = (int8_t*)d_ws;
    int8_t* w8 = x8 + (size_t)DIM_M * DIM_K;
    float*  s2 = (float*)(w8 + (size_t)DIM_N * DIM_K);
    float*  s1 = s2 + DIM_M;

    quant_rows_kernel<<<DIM_M, 256, 0, stream>>>(x, x8, s2);
    quant_rows_kernel<<<DIM_N, 256, 0, stream>>>(w, w8, s1);

    dim3 grid(DIM_N / TN, DIM_M / TM);
    igemm_wmma_kernel<<<grid, 256, 0, stream>>>(x8, w8, s2, s1, bias, out);
}
